// Encoder_27367531610364
// MI455X (gfx1250) — matrix operand order repaired and checked by reading, not compiled, not run on hardware
//
#include <hip/hip_runtime.h>
#include <cstdint>
#include <cstddef>

// ---------------- problem dims (from reference) ----------------
#define DIM      1024
#define HEADS    16
#define DIM_HEAD 64
#define INNER    1024        // HEADS * DIM_HEAD
#define DEPTH    3
#define FF       4096
#define BATCH    64
#define SEQ_N    128
#define SEQ_M    64
#define TOK_X    (BATCH * SEQ_N)   // 8192 query tokens
#define TOK_C    (BATCH * SEQ_M)   // 4096 kv tokens (cross)

// ---------------- CDNA5 WMMA types ----------------
typedef __attribute__((ext_vector_type(16))) __bf16 v16bf;
typedef __attribute__((ext_vector_type(8)))  __bf16 v8bf;
typedef __attribute__((ext_vector_type(8)))  float  v8f;
typedef __attribute__((ext_vector_type(4)))  unsigned int v4u;
typedef __attribute__((ext_vector_type(8)))  int v8i;
typedef __attribute__((ext_vector_type(4)))  int v4i;

union FragBF { v16bf v; v8bf half8[2]; };

// ---- Tensor Data Mover availability (arity differs across toolchains) ----
#if __has_builtin(__builtin_amdgcn_tensor_load_to_lds) && \
    __has_builtin(__builtin_amdgcn_s_wait_tensorcnt)
#define HAVE_TDM 1
#else
#define HAVE_TDM 0
#endif

// bf16 <-> f32 helpers on raw ushort storage (round-to-nearest-even)
__device__ __forceinline__ unsigned short f2bf(float f) {
  unsigned u = __float_as_uint(f);
  u += 0x7fffu + ((u >> 16) & 1u);
  return (unsigned short)(u >> 16);
}
__device__ __forceinline__ float bf2f(unsigned short s) {
  return __uint_as_float(((unsigned)s) << 16);
}

#if HAVE_TDM
// Issue a TDM 2D tile load: 128 rows x 32 bf16 from row-major [.,K] tensor
// into LDS at ldsOff, with 16B pad after each 64B row (=> 80B LDS row stride).
__device__ __forceinline__ void tdm_load_tile_a(const unsigned short* gsrc,
                                                unsigned ldsOff, int K) {
  unsigned long long ga = (unsigned long long)(uintptr_t)gsrc;  // byte address
  v4u g0;
  g0[0] = 1u;                                            // count=1 (valid D#)
  g0[1] = ldsOff;                                        // lds_addr (bytes)
  g0[2] = (unsigned)(ga & 0xffffffffull);                // global_addr[31:0]
  g0[3] = (unsigned)((ga >> 32) & 0x1ffffffull)          // global_addr[56:32]
        | (2u << 30);                                    // type = 2 ("image")
  v8i g1;
  g1[0] = (int)((1u << 16)     // data_size = 2 bytes
              | (1u << 20)     // pad_enable
              | (3u << 22)     // pad_interval: 16 dwords (64B) between pads
              | (3u << 25));   // pad_amount: 4 dwords (16B)
  g1[1] = (int)(((unsigned)K & 0xffffu) << 16);          // tensor_dim0 lo16
  g1[2] = (int)(((unsigned)K >> 16) | (128u << 16));     // dim0 hi16 | tensor_dim1=128
  g1[3] = (int)(32u << 16);                              // tile_dim0 = 32
  g1[4] = 128;                                           // tile_dim1 = 128, tile_dim2=0
  g1[5] = K;                                             // tensor_dim0_stride lo32
  g1[6] = 0;
  g1[7] = 0;
  v4i z4 = {0, 0, 0, 0};
#if __clang_major__ >= 23
  v8i z8 = {0, 0, 0, 0, 0, 0, 0, 0};
  __builtin_amdgcn_tensor_load_to_lds(g0, g1, z4, z4, z8, 0);
#else
  __builtin_amdgcn_tensor_load_to_lds(g0, g1, z4, z4, 0);
#endif
}
#endif

// ---------------- elementwise utility kernels ----------------
__global__ __launch_bounds__(256) void f32_to_bf16_kernel(
    const float* __restrict__ in, unsigned short* __restrict__ out, size_t n) {
  size_t i = (size_t)blockIdx.x * 256 + threadIdx.x;
  size_t stride = (size_t)gridDim.x * 256;
  for (; i < n; i += stride) out[i] = f2bf(in[i]);
}

__global__ __launch_bounds__(256) void f32_copy_kernel(
    const float* __restrict__ in, float* __restrict__ out, size_t n) {
  size_t i = (size_t)blockIdx.x * 256 + threadIdx.x;
  size_t stride = (size_t)gridDim.x * 256;
  for (; i < n; i += stride) out[i] = in[i];
}

// ---------------- RMSNorm ----------------
template <bool OUT_BF16>
__global__ __launch_bounds__(256) void rmsnorm_kernel(
    const float* __restrict__ X, const float* __restrict__ g,
    unsigned short* __restrict__ outb, float* __restrict__ outf) {
  __shared__ float red[8];
  const int row = blockIdx.x;
  const float* xr = X + (size_t)row * DIM;
  float ss = 0.f;
  for (int c = threadIdx.x; c < DIM; c += 256) { float v = xr[c]; ss += v * v; }
  for (int off = 16; off; off >>= 1) ss += __shfl_xor(ss, off, 32);
  const int wave = threadIdx.x >> 5, lane = threadIdx.x & 31;
  if (lane == 0) red[wave] = ss;
  __syncthreads();
  if (threadIdx.x == 0) {
    float t = 0.f;
    for (int w = 0; w < 8; w++) t += red[w];
    red[0] = t;
  }
  __syncthreads();
  const float nrm = sqrtf(red[0]) * 0.03125f;   // * DIM^-0.5
  const float scale = 1.0f / fmaxf(nrm, 1e-8f);
  for (int c = threadIdx.x; c < DIM; c += 256) {
    float v = xr[c] * scale * g[c];
    if constexpr (OUT_BF16) outb[(size_t)row * DIM + c] = f2bf(v);
    else                    outf[(size_t)row * DIM + c] = v;
  }
}

// ---------------- bf16 WMMA GEMM (double-buffered LDS, TDM A-staging) ----------------
// C[T, Ncols] = A[T, K] @ Bw[K, Ncols]   [+ bias] [gelu] [+ residual X (f32)]
// WG tile 128x128, 8 waves (2x4), wave tile 64x32 = 4x2 WMMA tiles, K-step 32.
template <bool BIAS, bool GELU, bool RESID>
__global__ __launch_bounds__(256) void gemm_bf16_wmma(
    const unsigned short* __restrict__ A,
    const unsigned short* __restrict__ Bw,
    const float* __restrict__ bias,
    float* __restrict__ Xres,
    unsigned short* __restrict__ OutB,
    int K, int Ncols) {
  // row stride 40 ushorts = 80B (multiple of 16B) -> aligned ds_load_b128
  __shared__ __align__(16) unsigned short Alds[2][128][40];  // [m][k]
  __shared__ __align__(16) unsigned short Blds[2][128][40];  // [n][k] (transposed)

  const int rowBase = blockIdx.y * 128;
  const int colBase = blockIdx.x * 128;
  const int tid  = threadIdx.x;
  const int wave = tid >> 5, lane = tid & 31;
  const int wm = wave >> 2, wn = wave & 3;
  const int hg = lane >> 4, lm = lane & 15;

  // per-thread staging coordinates
  const int ar = tid >> 1, ah = tid & 1;          // A: row, 16-elem half
  const int bk = tid >> 3, bn = tid & 7;          // B: k-row, 16-col group

  v8f acc[4][2] = {};

  // ---- prologue: stage k0 = 0 into buffer 0 (plain loads) ----
  {
    const v8bf* gp = (const v8bf*)(A + (size_t)(rowBase + ar) * K + 16 * ah);
    v8bf* lp = (v8bf*)(&Alds[0][ar][16 * ah]);
    lp[0] = gp[0];
    lp[1] = gp[1];
    const unsigned short* gb = Bw + (size_t)bk * Ncols + colBase + bn * 16;
#pragma unroll
    for (int e = 0; e < 16; e++) Blds[0][bn * 16 + e][bk] = gb[e];
  }
  __syncthreads();

  int buf = 0;
  for (int k0 = 0; k0 < K; k0 += 32) {
    const int nk = k0 + 32;
    const bool more = nk < K;

    // ---- kick off next tile: TDM DMA for A, register prefetch for B ----
#if !HAVE_TDM
    v8bf pa0, pa1;
#endif
    unsigned short pbv[16];
    if (more) {
#if HAVE_TDM
      if (wave == 0)
        tdm_load_tile_a(A + (size_t)rowBase * K + nk,
                        (unsigned)(uintptr_t)&Alds[buf ^ 1][0][0], K);
#else
      const v8bf* gp = (const v8bf*)(A + (size_t)(rowBase + ar) * K + nk + 16 * ah);
      pa0 = gp[0];
      pa1 = gp[1];
#endif
      const unsigned short* gb = Bw + (size_t)(nk + bk) * Ncols + colBase + bn * 16;
#pragma unroll
      for (int e = 0; e < 16; e++) pbv[e] = gb[e];
      if (nk + 32 < K)  // soft prefetch two tiles ahead -> global_prefetch_b8
        __builtin_prefetch(Bw + (size_t)(nk + 32 + bk) * Ncols + colBase + bn * 16, 0, 0);
    }

    // ---- compute from current buffer ----
    // A 16x32 frag: lane(h,m) elem e -> K = 16h + e
    FragBF a[4];
#pragma unroll
    for (int ms = 0; ms < 4; ms++) {
      const unsigned short* ap = &Alds[buf][wm * 64 + ms * 16 + lm][0];
      a[ms].half8[0] = *(const v8bf*)(ap + 16 * hg);
      a[ms].half8[1] = *(const v8bf*)(ap + 16 * hg + 8);
    }
    // B 32x16 frag: lane(h,n) elem e -> K = 16h + e (contiguous in transposed LDS)
    FragBF b[2];
#pragma unroll
    for (int ns = 0; ns < 2; ns++) {
      const unsigned short* bp = &Blds[buf][wn * 32 + ns * 16 + lm][0] + 16 * hg;
      b[ns].half8[0] = *(const v8bf*)(bp);
      b[ns].half8[1] = *(const v8bf*)(bp + 8);
    }
#pragma unroll
    for (int ms = 0; ms < 4; ms++)
#pragma unroll
      for (int ns = 0; ns < 2; ns++)
        acc[ms][ns] = __builtin_amdgcn_wmma_f32_16x16x32_bf16(
            false, a[ms].v, false, b[ns].v, (short)0, acc[ms][ns], false, false);

    // ---- commit next tile and synchronize ----
    if (more) {
#if !HAVE_TDM
      v8bf* lp = (v8bf*)(&Alds[buf ^ 1][ar][16 * ah]);
      lp[0] = pa0;
      lp[1] = pa1;
#endif
#pragma unroll
      for (int e = 0; e < 16; e++) Blds[buf ^ 1][bn * 16 + e][bk] = pbv[e];
#if HAVE_TDM
      if (wave == 0) __builtin_amdgcn_s_wait_tensorcnt(0);
#endif
      __syncthreads();
      buf ^= 1;
    }
  }

  // ---- epilogue: C layout M = 8*hg + r, N = lm ----
#pragma unroll
  for (int ms = 0; ms < 4; ms++) {
#pragma unroll
    for (int ns = 0; ns < 2; ns++) {
#pragma unroll
      for (int r = 0; r < 8; r++) {
        const int row = rowBase + wm * 64 + ms * 16 + hg * 8 + r;
        const int col = colBase + wn * 32 + ns * 16 + lm;
        float v = acc[ms][ns][r];
        if constexpr (BIAS) v += bias[col];
        if constexpr (GELU) v = 0.5f * v * (1.0f + erff(v * 0.70710678118654752f));
        const size_t idx = (size_t)row * Ncols + col;
        if constexpr (RESID) {
          v += Xres[idx];
          Xres[idx] = v;
        } else {
          OutB[idx] = f2bf(v);
        }
      }
    }
  }
}

// ---------------- WMMA attention (RoPE + softmax fused) ----------------
// NKV is compile-time so score accumulators never see dynamic indexing
// (round-2 disasm showed scratch spills with a runtime n-subtile count).
template <int NKV>
__global__ __launch_bounds__(256) void attention_wmma_kernel(
    const unsigned short* __restrict__ Q,
    const unsigned short* __restrict__ KV,
    unsigned short* __restrict__ Out,
    const unsigned char* __restrict__ mask,
    int Nq) {
  constexpr int NSUB = NKV / 16;  // 8 self / 4 cross
  // layout (ushorts):
  //   Ql: [128][72] @ 0, Kl: [128][72] @ 9216, Vt: [64][136] @ 18432
  //   Pl: [8][16][136] aliases Ql/Kl (17408 <= 18432), valid after barrier
  __shared__ __align__(16) unsigned short smem[27136];  // 53 KB
  unsigned short (*Ql)[72]  = (unsigned short(*)[72])(smem);
  unsigned short (*Kl)[72]  = (unsigned short(*)[72])(smem + 9216);
  unsigned short (*Vt)[136] = (unsigned short(*)[136])(smem + 18432);

  const int b = blockIdx.y, hh = blockIdx.x;
  const int tid = threadIdx.x;
  const int wave = tid >> 5, lane = tid & 31;
  const int hg = lane >> 4, lm = lane & 15;
  const size_t kvld = 2 * INNER;
  const float LN1E4 = 9.210340371976184f;  // ln(10000)

  // ---- stage Q (rope + scale) ----
  for (int idx = tid; idx < Nq * 64; idx += 256) {
    const int m = idx >> 6, d = idx & 63;
    const size_t base = (size_t)(b * Nq + m) * INNER + hh * 64;
    float v = bf2f(Q[base + d]) * 0.125f;  // DIM_HEAD^-0.5
    if (d < 32) {
      const int di = d & 15;
      const float th = (float)m * __expf(-(float)(2 * di) * (LN1E4 / 32.0f));
      const float partner = bf2f(Q[base + ((d < 16) ? d + 16 : d - 16)]) * 0.125f;
      v = (d < 16) ? v * __cosf(th) - partner * __sinf(th)
                   : v * __cosf(th) + partner * __sinf(th);
    }
    Ql[m][d] = f2bf(v);
  }
  // ---- stage K (rope) and V (transposed) ----
  for (int idx = tid; idx < NKV * 64; idx += 256) {
    const int j = idx >> 6, d = idx & 63;
    const size_t base = (size_t)(b * NKV + j) * kvld + hh * 64;
    float v = bf2f(KV[base + d]);
    if (d < 32) {
      const int di = d & 15;
      const float th = (float)j * __expf(-(float)(2 * di) * (LN1E4 / 32.0f));
      const float partner = bf2f(KV[base + ((d < 16) ? d + 16 : d - 16)]);
      v = (d < 16) ? v * __cosf(th) - partner * __sinf(th)
                   : v * __cosf(th) + partner * __sinf(th);
    }
    Kl[j][d] = f2bf(v);
    Vt[d][j] = KV[base + INNER + d];
  }
  __syncthreads();

  const int row0 = wave * 16;  // Nq = 128 = 8 waves * 16 rows

  // ---- S = Qr @ Kr^T : K-dim 64 = two 32-steps ----
  v8f s_acc[NSUB] = {};
#pragma unroll
  for (int kk = 0; kk < 64; kk += 32) {
    FragBF a;
    const unsigned short* ap = &Ql[row0 + lm][kk];
    a.half8[0] = *(const v8bf*)(ap + 16 * hg);
    a.half8[1] = *(const v8bf*)(ap + 16 * hg + 8);
#pragma unroll
    for (int ns = 0; ns < NSUB; ns++) {
      FragBF kb;  // Kl rows are exactly the transposed-B layout for S
      const unsigned short* bp = &Kl[ns * 16 + lm][kk + 16 * hg];
      kb.half8[0] = *(const v8bf*)(bp);
      kb.half8[1] = *(const v8bf*)(bp + 8);
      s_acc[ns] = __builtin_amdgcn_wmma_f32_16x16x32_bf16(
          false, a.v, false, kb.v, (short)0, s_acc[ns], false, false);
    }
  }

  // ---- mask + softmax in C layout (row M = 8*hg + r, key j = 16*ns + lm) ----
  if (mask != nullptr) {
#pragma unroll
    for (int ns = 0; ns < NSUB; ns++) {
      if (!mask[b * Nq + ns * 16 + lm]) {
#pragma unroll
        for (int r = 0; r < 8; r++) s_acc[ns][r] = -3.4e38f;
      }
    }
  }
  float pmax[8], psum[8];
#pragma unroll
  for (int r = 0; r < 8; r++) pmax[r] = -3.4e38f;
#pragma unroll
  for (int ns = 0; ns < NSUB; ns++)
#pragma unroll
    for (int r = 0; r < 8; r++) pmax[r] = fmaxf(pmax[r], s_acc[ns][r]);
#pragma unroll
  for (int off = 8; off; off >>= 1)  // reduce within the 16-lane half-group
#pragma unroll
    for (int r = 0; r < 8; r++) pmax[r] = fmaxf(pmax[r], __shfl_xor(pmax[r], off, 32));
#pragma unroll
  for (int r = 0; r < 8; r++) psum[r] = 0.f;
#pragma unroll
  for (int ns = 0; ns < NSUB; ns++)
#pragma unroll
    for (int r = 0; r < 8; r++) {
      const float p = __expf(s_acc[ns][r] - pmax[r]);
      s_acc[ns][r] = p;
      psum[r] += p;
    }
#pragma unroll
  for (int off = 8; off; off >>= 1)
#pragma unroll
    for (int r = 0; r < 8; r++) psum[r] += __shfl_xor(psum[r], off, 32);

  // ---- spill P (bf16) into per-wave LDS slice aliasing dead Ql/Kl ----
  __syncthreads();  // all waves done reading Ql/Kl
  unsigned short (*Pl)[136] = (unsigned short(*)[136])(smem + wave * 16 * 136);
#pragma unroll
  for (int ns = 0; ns < NSUB; ns++)
#pragma unroll
    for (int r = 0; r < 8; r++)
      Pl[8 * hg + r][ns * 16 + lm] = f2bf(s_acc[ns][r]);

  // ---- O = P @ V : K-dim = NKV, d-subtiles = 4 ----
  v8f o_acc[4] = {};
#pragma unroll
  for (int k0 = 0; k0 < NKV; k0 += 32) {
    FragBF pa;
    const unsigned short* ap = &Pl[lm][k0];
    pa.half8[0] = *(const v8bf*)(ap + 16 * hg);
    pa.half8[1] = *(const v8bf*)(ap + 16 * hg + 8);
#pragma unroll
    for (int dsb = 0; dsb < 4; dsb++) {
      FragBF vb;  // Vt rows ([d][j]) are the transposed-B layout for O
      const unsigned short* bp = &Vt[dsb * 16 + lm][k0 + 16 * hg];
      vb.half8[0] = *(const v8bf*)(bp);
      vb.half8[1] = *(const v8bf*)(bp + 8);
      o_acc[dsb] = __builtin_amdgcn_wmma_f32_16x16x32_bf16(
          false, pa.v, false, vb.v, (short)0, o_acc[dsb], false, false);
    }
  }

  // ---- epilogue: normalize by row sum, store bf16 ----
  float inv[8];
#pragma unroll
  for (int r = 0; r < 8; r++) inv[r] = 1.0f / psum[r];
#pragma unroll
  for (int dsb = 0; dsb < 4; dsb++) {
#pragma unroll
    for (int r = 0; r < 8; r++) {
      const int row = row0 + 8 * hg + r;
      const int d = dsb * 16 + lm;
      Out[(size_t)(b * Nq + row) * INNER + hh * 64 + d] = f2bf(o_acc[dsb][r] * inv[r]);
    }
  }
}

// ---------------- orchestration ----------------
extern "C" void kernel_launch(void* const* d_in, const int* in_sizes, int n_in,
                              void* d_out, int out_size, void* d_ws, size_t ws_size,
                              hipStream_t stream) {
  (void)in_sizes; (void)n_in; (void)out_size; (void)ws_size;
  const float* x_in    = (const float*)d_in[0];
  const float* cs_in   = (const float*)d_in[1];
  const unsigned char* mask = (const unsigned char*)d_in[2];
  const float* g_attn  = (const float*)d_in[3];
  const float* Wq_a    = (const float*)d_in[4];
  const float* Wkv_a   = (const float*)d_in[5];
  const float* Wo_a    = (const float*)d_in[6];
  const float* bo_a    = (const float*)d_in[7];
  const float* g_cross = (const float*)d_in[8];
  const float* Wq_c    = (const float*)d_in[9];
  const float* Wkv_c   = (const float*)d_in[10];
  const float* Wo_c    = (const float*)d_in[11];
  const float* bo_c    = (const float*)d_in[12];
  const float* g_ff    = (const float*)d_in[13];
  const float* W1      = (const float*)d_in[14];
  const float* b1      = (const float*)d_in[15];
  const float* W2      = (const float*)d_in[16];
  const float* b2      = (const float*)d_in[17];
  const float* g_outp  = (const float*)d_in[18];

  // workspace layout (~192 MB)
  char* w = (char*)d_ws;
  float* X            = (float*)w;          w += (size_t)TOK_X * DIM * 4;
  unsigned short* XN  = (unsigned short*)w; w += (size_t)TOK_X * DIM * 2;
  unsigned short* Qb  = (unsigned short*)w; w += (size_t)TOK_X * INNER * 2;
  unsigned short* KVb = (unsigned short*)w; w += (size_t)TOK_X * 2 * INNER * 2;
  unsigned short* AO  = (unsigned short*)w; w += (size_t)TOK_X * INNER * 2;
  unsigned short* Hb  = (unsigned short*)w; w += (size_t)TOK_X * FF * 2;
  unsigned short* CS  = (unsigned short*)w; w += (size_t)TOK_C * DIM * 2;
  unsigned short* WB  = (unsigned short*)w; w += (size_t)DIM * FF * 2;

  const int CG = 2048;

  f32_copy_kernel<<<CG, 256, 0, stream>>>(x_in, X, (size_t)TOK_X * DIM);
  f32_to_bf16_kernel<<<CG, 256, 0, stream>>>(cs_in, CS, (size_t)TOK_C * DIM);

  for (int i = 0; i < DEPTH; i++) {
    // ===== self-attention =====
    rmsnorm_kernel<true><<<TOK_X, 256, 0, stream>>>(X, g_attn + (size_t)i * DIM, XN, nullptr);

    f32_to_bf16_kernel<<<CG, 256, 0, stream>>>(Wq_a + (size_t)i * DIM * INNER, WB, (size_t)DIM * INNER);
    gemm_bf16_wmma<false, false, false><<<dim3(INNER / 128, TOK_X / 128), 256, 0, stream>>>(
        XN, WB, nullptr, nullptr, Qb, DIM, INNER);

    f32_to_bf16_kernel<<<CG, 256, 0, stream>>>(Wkv_a + (size_t)i * DIM * 2 * INNER, WB, (size_t)DIM * 2 * INNER);
    gemm_bf16_wmma<false, false, false><<<dim3(2 * INNER / 128, TOK_X / 128), 256, 0, stream>>>(
        XN, WB, nullptr, nullptr, KVb, DIM, 2 * INNER);

    attention_wmma_kernel<SEQ_N><<<dim3(HEADS, BATCH), 256, 0, stream>>>(Qb, KVb, AO, mask, SEQ_N);

    f32_to_bf16_kernel<<<CG, 256, 0, stream>>>(Wo_a + (size_t)i * INNER * DIM, WB, (size_t)INNER * DIM);
    gemm_bf16_wmma<true, false, true><<<dim3(DIM / 128, TOK_X / 128), 256, 0, stream>>>(
        AO, WB, bo_a + (size_t)i * DIM, X, nullptr, INNER, DIM);

    // ===== cross-attention (kv from raw chunked_seq, per reference) =====
    rmsnorm_kernel<true><<<TOK_X, 256, 0, stream>>>(X, g_cross + (size_t)i * DIM, XN, nullptr);

    f32_to_bf16_kernel<<<CG, 256, 0, stream>>>(Wq_c + (size_t)i * DIM * INNER, WB, (size_t)DIM * INNER);
    gemm_bf16_wmma<false, false, false><<<dim3(INNER / 128, TOK_X / 128), 256, 0, stream>>>(
        XN, WB, nullptr, nullptr, Qb, DIM, INNER);

    f32_to_bf16_kernel<<<CG, 256, 0, stream>>>(Wkv_c + (size_t)i * DIM * 2 * INNER, WB, (size_t)DIM * 2 * INNER);
    gemm_bf16_wmma<false, false, false><<<dim3(2 * INNER / 128, TOK_C / 128), 256, 0, stream>>>(
        CS, WB, nullptr, nullptr, KVb, DIM, 2 * INNER);

    attention_wmma_kernel<SEQ_M><<<dim3(HEADS, BATCH), 256, 0, stream>>>(Qb, KVb, AO, nullptr, SEQ_N);

    f32_to_bf16_kernel<<<CG, 256, 0, stream>>>(Wo_c + (size_t)i * INNER * DIM, WB, (size_t)INNER * DIM);
    gemm_bf16_wmma<true, false, true><<<dim3(DIM / 128, TOK_X / 128), 256, 0, stream>>>(
        AO, WB, bo_c + (size_t)i * DIM, X, nullptr, INNER, DIM);

    // ===== feed-forward =====
    rmsnorm_kernel<true><<<TOK_X, 256, 0, stream>>>(X, g_ff + (size_t)i * DIM, XN, nullptr);

    f32_to_bf16_kernel<<<CG, 256, 0, stream>>>(W1 + (size_t)i * DIM * FF, WB, (size_t)DIM * FF);
    gemm_bf16_wmma<true, true, false><<<dim3(FF / 128, TOK_X / 128), 256, 0, stream>>>(
        XN, WB, b1 + (size_t)i * FF, nullptr, Hb, DIM, FF);

    f32_to_bf16_kernel<<<CG, 256, 0, stream>>>(W2 + (size_t)i * FF * DIM, WB, (size_t)FF * DIM);
    gemm_bf16_wmma<true, false, true><<<dim3(DIM / 128, TOK_X / 128), 256, 0, stream>>>(
        Hb, WB, b2 + (size_t)i * DIM, X, nullptr, FF, DIM);
  }

  rmsnorm_kernel<false><<<TOK_X, 256, 0, stream>>>(X, g_outp, nullptr, (float*)d_out);
}
